// tDBN_1_55044300866040
// MI455X (gfx1250) — matrix operand order repaired and checked by reading, not compiled, not run on hardware
//
#include <hip/hip_runtime.h>
#include <hip/hip_bf16.h>
#include <stdint.h>

// ---------------------------------------------------------------------------
// CDNA5 (gfx1250) implicit-GEMM 3D conv network using v_wmma_f32_16x16x32_bf16
// Round 4: fully branchless staging (pin folded into coords, bitwise predicate,
//          integer cndmask zero-fill); 4 WMMA / K-step; LDS double buffering.
// ---------------------------------------------------------------------------

typedef __attribute__((ext_vector_type(16))) __bf16 v16bf;
typedef __attribute__((ext_vector_type(8)))  float  v8f;

#define BN_EPS 1e-3f

// ---- fold BN params into (g, b): y = relu(x*g + b) ------------------------
__global__ void bn_fold_kernel(const float* __restrict__ scale,
                               const float* __restrict__ bias,
                               const float* __restrict__ mean,
                               const float* __restrict__ var,
                               float* __restrict__ gb, int C)
{
    int c = blockIdx.x * blockDim.x + threadIdx.x;
    if (c < C) {
        float g = scale[c] * rsqrtf(var[c] + BN_EPS);
        gb[c]     = g;
        gb[C + c] = bias[c] - mean[c] * g;
    }
}

// ---- scatter voxel features into dense bf16 grid + occupancy mask ---------
__global__ void scatter_kernel(const float* __restrict__ feats,
                               const int* __restrict__ coors,
                               __bf16* __restrict__ x0,
                               float* __restrict__ mask, int N)
{
    int idx = blockIdx.x * blockDim.x + threadIdx.x;
    if (idx >= N * 128) return;
    int i = idx >> 7, c = idx & 127;
    int d = coors[i * 4 + 1], h = coors[i * 4 + 2], w = coors[i * 4 + 3];
    int off = (d * 63 + h) * 63 + w;
    x0[(size_t)c * (31 * 63 * 63) + off] = (__bf16)feats[idx];
    if (c == 0) mask[off] = 1.0f;
}

// ---- VALID max pool on the {0,1} mask (reduce init = 0 like reference) ----
__global__ void maxpool_kernel(const float* __restrict__ in, float* __restrict__ out,
                               int Di, int Hi, int Wi,
                               int kd, int kh, int kw,
                               int sd, int sh, int sw,
                               int Do, int Ho, int Wo)
{
    int idx = blockIdx.x * blockDim.x + threadIdx.x;
    int P = Do * Ho * Wo;
    if (idx >= P) return;
    int od = idx / (Ho * Wo);
    int r = idx - od * Ho * Wo;
    int oh = r / Wo;
    int ow = r - oh * Wo;
    float m = 0.0f;
    for (int z = 0; z < kd; ++z)
        for (int y = 0; y < kh; ++y)
            for (int x = 0; x < kw; ++x) {
                int id = od * sd + z, ih = oh * sh + y, iw = ow * sw + x;
                float v = in[(id * Hi + ih) * Wi + iw];
                m = v > m ? v : m;
            }
    out[idx] = m;
}

// ---- pre-pack f32 OIDHW weights into bf16 A-operand lane order ------------
// Layout: wp[((kt*ctiles + ct)*32 + lane)*16 + e]
// This code's contiguous K order, not the ISA table order (A 16x32, B matches): lane<16 holds row M=lane, K={0..15};
// lane>=16 holds row M=lane-16, K={16..31}; 2 halfs per VGPR.
__global__ void prepack_kernel(const float* __restrict__ w, __bf16* __restrict__ wp,
                               int Cout, int Ktot)
{
    int idx = blockIdx.x * blockDim.x + threadIdx.x;
    int total = (Ktot >> 5) * (Cout >> 4) * 512;
    if (idx >= total) return;
    int e    = idx & 15;
    int lane = (idx >> 4) & 31;
    int rest = idx >> 9;
    int ctiles = Cout >> 4;
    int ct = rest % ctiles;
    int kt = rest / ctiles;
    int m = lane & 15, hi = lane >> 4;
    int v = e >> 1, pb = e & 1;
    int kk = hi * 16 + v * 2 + pb;
    int k  = (kt << 5) + kk;
    int co = (ct << 4) + m;
    wp[idx] = (__bf16)w[(size_t)co * Ktot + k];
}

// ---- per-layer k-decomposition table: kills inner-loop integer division ----
// ktab[k].x = ci*DHWi + z*HiWi + y*Wi + x  (input offset, before output base)
// ktab[k].y = z | y<<8 | x<<16             (for bounds checks)
__global__ void ktab_kernel(int2* __restrict__ ktab, int Cin,
                            int kd, int kh, int kw, int Wi, int HiWi, int DHWi)
{
    int k = blockIdx.x * blockDim.x + threadIdx.x;
    int khw = kh * kw, kdhw = kd * khw;
    int Ktot = Cin * kdhw;
    if (k >= Ktot) return;
    int ci = k / kdhw;  int r  = k - ci * kdhw;
    int z  = r / khw;   int r2 = r - z * khw;
    int y  = r2 / kw;   int x  = r2 - y * kw;
    int2 t;
    t.x = ci * DHWi + z * HiWi + y * Wi + x;
    t.y = z | (y << 8) | (x << 16);
    ktab[k] = t;
}

// ---- fused implicit-GEMM conv3d + BN + ReLU + mask ------------------------
// Block = 128 threads = 4 waves. Block tile: 64 output positions x 64 cout.
// Each wave: one 16-cout A tile, four 16x16 accumulators (4 WMMA / K-step).
__global__ __launch_bounds__(128)
void conv3d_wmma_kernel(const __bf16* __restrict__ in,
                        int Di, int Hi, int Wi,
                        const __bf16* __restrict__ wp, int Cout,
                        const int2* __restrict__ ktab, int Ktot,
                        int kdm1, int khm1, int kwm1,
                        int sd, int sh, int sw,
                        int pd, int ph, int pw,
                        const float* __restrict__ gb,
                        const float* __restrict__ mask,
                        __bf16* __restrict__ outb,
                        float* __restrict__ outf,
                        int Do, int Ho, int Wo)
{
    const int P      = Do * Ho * Wo;
    const int pbase  = blockIdx.x * 64;
    const int HiWi   = Hi * Wi;
    const int nkt    = Ktot >> 5;          // Ktot multiple of 32 for every layer
    const int ctiles = Cout >> 4;
    const int wave   = threadIdx.x >> 5;
    const int lane   = threadIdx.x & 31;
    const int ct     = blockIdx.y * 4 + wave;

    __shared__ __align__(32) __bf16 xt[2][64 * 32];   // double-buffered [N=64][K=32]

    // staging role: this thread fills 16 consecutive K for one N
    const int n  = threadIdx.x >> 1;         // 0..63
    const int kq = (threadIdx.x & 1) * 16;   // 0 or 16
    const int p  = pbase + n;
    const bool pin = p < P;
    int od = 0, oh = 0, ow = 0;
    if (pin) {
        od = p / (Ho * Wo);
        int rem = p - od * Ho * Wo;
        oh = rem / Wo;
        ow = rem - oh * Wo;
    }
    // fold the p<P predicate into the coordinates: out-of-range threads get
    // coordinates that fail every bounds check arithmetically (no branch).
    const int OOB  = -(1 << 28);
    const int idb  = pin ? od * sd - pd : OOB;
    const int ihb  = pin ? oh * sh - ph : OOB;
    const int iwb  = pin ? ow * sw - pw : OOB;
    const int base = idb * HiWi + ihb * Wi + iwb;
    // fast path: full kernel footprint in-bounds for this output position
    const bool interior =
        idb >= 0 && (idb + kdm1) < Di &&
        ihb >= 0 && (ihb + khm1) < Hi &&
        iwb >= 0 && (iwb + kwm1) < Wi;

    const unsigned short* inu = (const unsigned short*)in;
    const int bn  = lane & 15;
    const int bhi = lane >> 4;

    v8f acc[4] = {{}, {}, {}, {}};
    for (int kt = 0; kt < nkt; ++kt) {
        // ---- stage B tile (division-free, branchless im2col gather) ----
        unsigned short* dst = (unsigned short*)&xt[kt & 1][n * 32 + kq];
        const int2* tp      = ktab + (kt << 5) + kq;
        if (interior) {
#pragma unroll
            for (int j = 0; j < 16; ++j) {
                int2 t = tp[j];
                dst[j] = inu[t.x + base];
            }
        } else {
#pragma unroll
            for (int j = 0; j < 16; ++j) {
                int2 t = tp[j];
                int z = t.y & 255, y = (t.y >> 8) & 255, x = t.y >> 16;
                int ok = ((unsigned)(idb + z) < (unsigned)Di) &
                         ((unsigned)(ihb + y) < (unsigned)Hi) &
                         ((unsigned)(iwb + x) < (unsigned)Wi);
                int addr = ok ? (t.x + base) : 0;    // clamp: in[0] always valid
                unsigned short v = inu[addr];
                dst[j] = ok ? v : (unsigned short)0;
            }
        }
        __syncthreads();   // single barrier per K-step (ping-pong buffers)

        // ---- A fragment: packed weights, contiguous 32B per lane ----
        size_t abase = ((size_t)(kt * ctiles + ct) * 32 + lane) * 16;
        v16bf afrag = *(const v16bf*)(wp + abase);
        if (kt + 1 < nkt)
            __builtin_prefetch(wp + abase + (size_t)ctiles * 512, 0, 1);

        // ---- four B fragments from LDS, four WMMAs sharing one A ----
        const __bf16* xb = xt[kt & 1];
#pragma unroll
        for (int g = 0; g < 4; ++g) {
            v16bf bf = *(const v16bf*)(xb + (g * 16 + bn) * 32 + bhi * 16);
            acc[g] = __builtin_amdgcn_wmma_f32_16x16x32_bf16(
                false, afrag, false, bf, (short)0, acc[g], false, false);
        }
    }

    // ---- epilogue: BN + ReLU + mask, write bf16 and/or f32 ----
#pragma unroll
    for (int g = 0; g < 4; ++g) {
        int pp = pbase + g * 16 + bn;
        if (pp < P) {
            float m = mask ? mask[pp] : 1.0f;
#pragma unroll
            for (int r = 0; r < 8; ++r) {
                int co = (ct << 4) + (bhi << 3) + r;
                float vv = acc[g][r] * gb[co] + gb[Cout + co];
                vv = vv > 0.0f ? vv : 0.0f;
                vv *= m;
                if (outb) outb[(size_t)co * P + pp] = (__bf16)vv;
                if (outf) outf[(size_t)co * P + pp] = vv;
            }
        }
    }
}

// ---------------------------------------------------------------------------
// Host orchestration
// ---------------------------------------------------------------------------
extern "C" void kernel_launch(void* const* d_in, const int* in_sizes, int n_in,
                              void* d_out, int out_size, void* d_ws, size_t ws_size,
                              hipStream_t stream)
{
    (void)n_in; (void)out_size; (void)ws_size;
    const float* vf    = (const float*)d_in[0];
    const int*   coors = (const int*)d_in[1];
    int N = in_sizes[0] / 128;

    // ---- workspace layout ----
    char* ws = (char*)d_ws;
    size_t off = 0;
    auto alloc = [&](size_t bytes) -> char* {
        char* pp = ws + off;
        off = (off + bytes + 255) & ~(size_t)255;
        return pp;
    };
    __bf16* X0    = (__bf16*)alloc(sizeof(__bf16) * 128UL * 123039);  // dense input grid
    float*  MASK0 = (float*) alloc(sizeof(float) * 123039);
    __bf16* ACTA  = (__bf16*)alloc(sizeof(__bf16) * 8388608UL);       // ping
    __bf16* ACTB  = (__bf16*)alloc(sizeof(__bf16) * 8388608UL);       // pong
    __bf16* WP    = (__bf16*)alloc(sizeof(__bf16) * 8388608UL);       // packed weights scratch
    int2*   KTAB  = (int2*)  alloc(sizeof(int2) * 13824);             // max Ktot
    float*  GB    = (float*) alloc(sizeof(float) * 13 * 1024);        // folded BN per layer
    float*  MK0   = (float*) alloc(sizeof(float) * 35721);            // 9*63*63
    float*  M1    = (float*) alloc(sizeof(float) * 14415);            // 15*31*31
    float*  MK1   = (float*) alloc(sizeof(float) * 8649);             // 9*31*31
    float*  M2    = (float*) alloc(sizeof(float) * 1575);             // 7*15*15
    float*  MK2   = (float*) alloc(sizeof(float) * 1125);             // 5*15*15
    float*  M3    = (float*) alloc(sizeof(float) * 147);              // 3*7*7

    hipMemsetAsync(X0, 0, sizeof(__bf16) * 128UL * 123039, stream);
    hipMemsetAsync(MASK0, 0, sizeof(float) * 123039, stream);

    scatter_kernel<<<(N * 128 + 255) / 256, 256, 0, stream>>>(vf, coors, X0, MASK0, N);

    // ---- fold BN params for all 13 conv layers ----
    // d_in: 0 vf, 1 coors, 2 batch; then per layer: w, scale, bias, mean, var
    const int wi[13] = {3, 8, 13, 18, 23, 28, 33, 38, 43, 48, 53, 58, 63};
    const int cc[13] = {64, 64, 128, 128, 128, 128, 256, 256, 256, 256, 512, 512, 512};
    for (int l = 0; l < 13; ++l) {
        bn_fold_kernel<<<(cc[l] + 255) / 256, 256, 0, stream>>>(
            (const float*)d_in[wi[l] + 1], (const float*)d_in[wi[l] + 2],
            (const float*)d_in[wi[l] + 3], (const float*)d_in[wi[l] + 4],
            GB + l * 1024, cc[l]);
    }

    auto pool = [&](const float* pin, int Di, int Hi, int Wi,
                    int kd, int kh, int kw, int sd, int sh, int sw,
                    float* pout, int Do, int Ho, int Wo) {
        int P = Do * Ho * Wo;
        maxpool_kernel<<<(P + 255) / 256, 256, 0, stream>>>(
            pin, pout, Di, Hi, Wi, kd, kh, kw, sd, sh, sw, Do, Ho, Wo);
    };
    pool(MASK0, 31, 63, 63, 15, 1, 1, 2, 1, 1, MK0, 9, 63, 63);
    pool(MASK0, 31, 63, 63, 3, 3, 3, 2, 2, 2, M1, 15, 31, 31);
    pool(M1, 15, 31, 31, 7, 1, 1, 1, 1, 1, MK1, 9, 31, 31);
    pool(M1, 15, 31, 31, 3, 3, 3, 2, 2, 2, M2, 7, 15, 15);
    pool(M2, 7, 15, 15, 3, 1, 1, 1, 1, 1, MK2, 5, 15, 15);
    pool(M2, 7, 15, 15, 3, 3, 3, 2, 2, 2, M3, 3, 7, 7);

    float* OUT = (float*)d_out;
    float* Y0 = OUT;                 // 64*9*63*63   = 2286144
    float* Y1 = Y0 + 2286144;        // 128*9*31*31  = 1107072
    float* Y2 = Y1 + 1107072;        // 256*5*15*15  = 288000
    float* Y3 = Y2 + 288000;         // 512*3*7*7    = 75264

    auto conv = [&](const __bf16* cin, int Cin, int Di, int Hi, int Wi,
                    const float* w, int Cout, int kd, int kh, int kw,
                    int sd, int sh, int sw, int pd, int ph, int pw,
                    int li, const float* mask, __bf16* outb, float* outf,
                    int Do, int Ho, int Wo) {
        int Ktot = Cin * kd * kh * kw;
        int total = (Ktot / 32) * (Cout / 16) * 512;
        prepack_kernel<<<(total + 255) / 256, 256, 0, stream>>>(w, WP, Cout, Ktot);
        ktab_kernel<<<(Ktot + 255) / 256, 256, 0, stream>>>(
            KTAB, Cin, kd, kh, kw, Wi, Hi * Wi, Di * Hi * Wi);
        int P = Do * Ho * Wo;
        dim3 grid((P + 63) / 64, Cout / 64);
        conv3d_wmma_kernel<<<grid, 128, 0, stream>>>(
            cin, Di, Hi, Wi, WP, Cout, KTAB, Ktot,
            kd - 1, kh - 1, kw - 1, sd, sh, sw,
            pd, ph, pw, GB + li * 1024, mask, outb, outf, Do, Ho, Wo);
    };

    // b0: 128->64, 3^3, s1, p1, *mask
    conv(X0, 128, 31, 63, 63, (const float*)d_in[3], 64, 3, 3, 3, 1, 1, 1, 1, 1, 1,
         0, MASK0, ACTA, nullptr, 31, 63, 63);
    // fm0: 64->64, (15,1,1), s(2,1,1), VALID -> Y0
    conv(ACTA, 64, 31, 63, 63, (const float*)d_in[8], 64, 15, 1, 1, 2, 1, 1, 0, 0, 0,
         1, MK0, nullptr, Y0, 9, 63, 63);
    // block1
    conv(ACTA, 64, 31, 63, 63, (const float*)d_in[13], 128, 3, 3, 3, 2, 2, 2, 0, 0, 0,
         2, M1, ACTB, nullptr, 15, 31, 31);
    conv(ACTB, 128, 15, 31, 31, (const float*)d_in[18], 128, 3, 3, 3, 1, 1, 1, 1, 1, 1,
         3, M1, ACTA, nullptr, 15, 31, 31);
    conv(ACTA, 128, 15, 31, 31, (const float*)d_in[23], 128, 3, 3, 3, 1, 1, 1, 1, 1, 1,
         4, M1, ACTB, nullptr, 15, 31, 31);
    conv(ACTB, 128, 15, 31, 31, (const float*)d_in[28], 128, 7, 1, 1, 1, 1, 1, 0, 0, 0,
         5, MK1, nullptr, Y1, 9, 31, 31);
    // block2
    conv(ACTB, 128, 15, 31, 31, (const float*)d_in[33], 256, 3, 3, 3, 2, 2, 2, 0, 0, 0,
         6, M2, ACTA, nullptr, 7, 15, 15);
    conv(ACTA, 256, 7, 15, 15, (const float*)d_in[38], 256, 3, 3, 3, 1, 1, 1, 1, 1, 1,
         7, M2, ACTB, nullptr, 7, 15, 15);
    conv(ACTB, 256, 7, 15, 15, (const float*)d_in[43], 256, 3, 3, 3, 1, 1, 1, 1, 1, 1,
         8, M2, ACTA, nullptr, 7, 15, 15);
    conv(ACTA, 256, 7, 15, 15, (const float*)d_in[48], 256, 3, 1, 1, 1, 1, 1, 0, 0, 0,
         9, MK2, nullptr, Y2, 5, 15, 15);
    // block3
    conv(ACTA, 256, 7, 15, 15, (const float*)d_in[53], 512, 3, 3, 3, 2, 2, 2, 0, 0, 0,
         10, M3, ACTB, nullptr, 3, 7, 7);
    conv(ACTB, 512, 3, 7, 7, (const float*)d_in[58], 512, 3, 3, 3, 1, 1, 1, 1, 1, 1,
         11, M3, ACTA, nullptr, 3, 7, 7);
    conv(ACTA, 512, 3, 7, 7, (const float*)d_in[63], 512, 3, 3, 3, 1, 1, 1, 1, 1, 1,
         12, M3, nullptr, Y3, 3, 7, 7);
}
